// DeepSpeedSparseSelfAttention_90460601188604
// MI455X (gfx1250) — matrix operand order repaired and checked by reading, not compiled, not run on hardware
//
#include <hip/hip_runtime.h>
#include <hip/hip_bf16.h>

// Block-sparse flash attention for the DeepSpeed "fixed" layout.
// One wave32 per (block-row, batch*head). bf16 WMMA (16x16x32) with f32 accum.
// Softmax denominator computed on the WMMA pipe (P @ ones). V tile staged to
// LDS via GLOBAL_LOAD_ASYNC_TO_LDS (ASYNCcnt) when the builtin is available.

typedef __attribute__((ext_vector_type(16))) __bf16 v16bf;
typedef __attribute__((ext_vector_type(8)))  float  v8f;
typedef __attribute__((ext_vector_type(4)))  int    v4i;

#define AS1 __attribute__((address_space(1)))
#define AS3 __attribute__((address_space(3)))

#define B_   2
#define H_   16
#define L_   4096
#define HD_  64
#define BLK_ 32
#define NB_  (L_ / BLK_)   // 128

__device__ __forceinline__ float rowmax16(float v) {
#pragma unroll
  for (int m = 1; m < 16; m <<= 1) v = fmaxf(v, __shfl_xor(v, m, 32));
  return v;
}

__global__ __launch_bounds__(32)
void bsattn_wmma_bf16(const float* __restrict__ q,
                      const float* __restrict__ k,
                      const float* __restrict__ v,
                      const int*   __restrict__ rows,
                      const int*   __restrict__ cols,
                      float*       __restrict__ out,
                      int nnz)
{
  const int r    = blockIdx.x;           // block-row index (0..NB_-1)
  const int bh   = blockIdx.y;           // fused batch*head (0..31)
  const int lane = threadIdx.x;          // wave32
  const int hl   = lane >> 4;            // lane half (0/1)
  const int lm   = lane & 15;            // lane within half
  const float scale = 0.125f;            // 64^-0.5

  const long base = (long)bh * L_ * HD_;
  const float* qb = q   + base + (long)r * BLK_ * HD_;
  float*       ob = out + base + (long)r * BLK_ * HD_;

  // ---- range [s0, s1) of active blocks in row r (rows[] sorted by row) ----
  int lo = 0, hi = nnz;
  while (lo < hi) { int mid = (lo + hi) >> 1; if (rows[mid] < r)     lo = mid + 1; else hi = mid; }
  const int s0 = lo;
  hi = nnz;
  while (lo < hi) { int mid = (lo + hi) >> 1; if (rows[mid] < r + 1) lo = mid + 1; else hi = mid; }
  const int s1 = lo;

  // ---- preload Q block as bf16 A-fragments: [Mtile][kstep] ----
  // A 16x32 bf16 layout: lanes 0-15 hold K=0..7,16..23 ; lanes 16-31 hold K=8..15,24..31
  v16bf qA[2][2];
#pragma unroll
  for (int mt = 0; mt < 2; ++mt) {
#pragma unroll
    for (int ks = 0; ks < 2; ++ks) {
      const float* rp = qb + (mt * 16 + lm) * HD_ + ks * 32;
#pragma unroll
      for (int j = 0; j < 8; ++j) qA[mt][ks][j]     = (__bf16)rp[hl * 8 + j];
#pragma unroll
      for (int j = 0; j < 8; ++j) qA[mt][ks][8 + j] = (__bf16)rp[16 + hl * 8 + j];
    }
  }

  // constant all-ones B fragment: row-sum of P via the WMMA pipe
  v16bf onesB;
#pragma unroll
  for (int j = 0; j < 16; ++j) onesB[j] = (__bf16)1.0f;

  // ---- accumulators & online-softmax state ----
  v8f o[2][4];
#pragma unroll
  for (int mt = 0; mt < 2; ++mt)
#pragma unroll
    for (int dt = 0; dt < 4; ++dt) o[mt][dt] = (v8f){};

  float mst[2][8], lst[2][8];
#pragma unroll
  for (int mt = 0; mt < 2; ++mt)
#pragma unroll
    for (int i = 0; i < 8; ++i) { mst[mt][i] = -__builtin_inff(); lst[mt][i] = 0.0f; }

  __shared__ float  sVf[BLK_ * HD_];   // 32x64 V tile, f32 (8 KB) — async DMA target
  __shared__ __bf16 sP[BLK_ * BLK_];   // 32x32 P tile, bf16 (2 KB)

  for (int e = s0; e < s1; ++e) {
    const int c = cols[e];
    const float* kb = k + base + (long)c * BLK_ * HD_;
    const float* vb = v + base + (long)c * BLK_ * HD_;

    // prefetch next block's K into caches (global_prefetch_b8)
    if (e + 1 < s1) {
      const int c2 = cols[e + 1];
      __builtin_prefetch(k + base + (long)c2 * BLK_ * HD_ + lane * HD_, 0, 1);
    }

    // Async LDS writes are NOT ordered with normal ds reads: make sure the
    // previous iteration's V-fragment ds loads have completed first.
    asm volatile("s_wait_dscnt 0x0" ::: "memory");

    // ---- stage V block (contiguous 32x64 f32) into LDS ----
#if __has_builtin(__builtin_amdgcn_global_load_async_to_lds_b128)
#pragma unroll
    for (int t = 0; t < 16; ++t) {
      const int idx = t * 32 + lane;                 // float4 / int4 index
      __builtin_amdgcn_global_load_async_to_lds_b128(
          (AS1 v4i*)(vb + idx * 4), (AS3 v4i*)&sVf[idx * 4], 0, 0);
    }
#else
#pragma unroll
    for (int t = 0; t < 16; ++t) {
      const int idx = t * 32 + lane;                 // float4 index
      ((float4*)sVf)[idx] = ((const float4*)vb)[idx];
    }
#endif

    // ---- S = (Q @ K^T) : 2x2 tiles of 16x16, K-depth 64 in two 32-steps ----
    v8f s[2][2];
#pragma unroll
    for (int mt = 0; mt < 2; ++mt)
#pragma unroll
      for (int nt = 0; nt < 2; ++nt) s[mt][nt] = (v8f){};

#pragma unroll
    for (int ks = 0; ks < 2; ++ks) {
      v16bf kB[2];
      // B 32x16 bf16 layout: lanes 0-15 hold K=0..7,16..23 ; lanes 16-31 hold K=8..15,24..31; N = lane&15
#pragma unroll
      for (int nt = 0; nt < 2; ++nt) {
        const float* rp = kb + (nt * 16 + lm) * HD_ + ks * 32 + hl * 8;
#pragma unroll
        for (int j = 0; j < 16; ++j) kB[nt][j] = (__bf16)rp[j < 8 ? j : 8 + j];
      }
#pragma unroll
      for (int mt = 0; mt < 2; ++mt)
#pragma unroll
        for (int nt = 0; nt < 2; ++nt)
          s[mt][nt] = __builtin_amdgcn_wmma_f32_16x16x32_bf16(
              false, qA[mt][ks], false, kB[nt], (short)0, s[mt][nt], false, false);
    }

    // ---- online softmax; C-tile layout: element i -> row hl*8+i, col lm ----
    float al[2][8];
#pragma unroll
    for (int mt = 0; mt < 2; ++mt) {
      float p0[8], p1[8];
#pragma unroll
      for (int i = 0; i < 8; ++i) {
        const float a  = s[mt][0][i] * scale;
        const float bb = s[mt][1][i] * scale;
        const float mx = rowmax16(fmaxf(a, bb));
        const float mn = fmaxf(mst[mt][i], mx);
        al[mt][i] = __expf(mst[mt][i] - mn);
        mst[mt][i] = mn;
        p0[i] = __expf(a  - mn);
        p1[i] = __expf(bb - mn);
#pragma unroll
        for (int dt = 0; dt < 4; ++dt) o[mt][dt][i] *= al[mt][i];
      }
      // write P tile (bf16) to LDS, row-major 32 cols
#pragma unroll
      for (int i = 0; i < 8; ++i) {
        const int row = mt * 16 + hl * 8 + i;
        sP[row * BLK_ + lm]      = (__bf16)p0[i];
        sP[row * BLK_ + 16 + lm] = (__bf16)p1[i];
      }
    }

    // ---- P as A-frag (via LDS transpose) ----
    v16bf pA[2];
#pragma unroll
    for (int mt = 0; mt < 2; ++mt) {
      const int rbase = (mt * 16 + lm) * BLK_;
#pragma unroll
      for (int j = 0; j < 8; ++j) pA[mt][j]     = sP[rbase + hl * 8 + j];
#pragma unroll
      for (int j = 0; j < 8; ++j) pA[mt][8 + j] = sP[rbase + 16 + hl * 8 + j];
    }

    // ---- softmax denominator on the WMMA pipe: rowsum(P) = P @ ones ----
#pragma unroll
    for (int mt = 0; mt < 2; ++mt) {
      v8f z = (v8f){};
      z = __builtin_amdgcn_wmma_f32_16x16x32_bf16(
          false, pA[mt], false, onesB, (short)0, z, false, false);
#pragma unroll
      for (int i = 0; i < 8; ++i)
        lst[mt][i] = lst[mt][i] * al[mt][i] + z[i];
    }

    // ---- wait for the async V DMA, then O += P @ V ----
    asm volatile("s_wait_asynccnt 0x0" ::: "memory");

#pragma unroll
    for (int dt = 0; dt < 4; ++dt) {
      v16bf vB;
      // B 32x16: lanes 0-15 hold K=0..7,16..23 ; lanes 16-31 hold K=8..15,24..31; N = lane&15
#pragma unroll
      for (int j = 0; j < 16; ++j)
        vB[j] = (__bf16)sVf[(hl * 8 + (j < 8 ? j : 8 + j)) * HD_ + dt * 16 + lm];
#pragma unroll
      for (int mt = 0; mt < 2; ++mt)
        o[mt][dt] = __builtin_amdgcn_wmma_f32_16x16x32_bf16(
            false, pA[mt], false, vB, (short)0, o[mt][dt], false, false);
    }
  }

  // ---- normalize and store f32 output ----
#pragma unroll
  for (int mt = 0; mt < 2; ++mt) {
#pragma unroll
    for (int i = 0; i < 8; ++i) {
      const float inv = (lst[mt][i] > 0.0f) ? (1.0f / lst[mt][i]) : 0.0f;
      const int row = mt * 16 + hl * 8 + i;
#pragma unroll
      for (int dt = 0; dt < 4; ++dt)
        ob[row * HD_ + dt * 16 + lm] = o[mt][dt][i] * inv;
    }
  }
}

extern "C" void kernel_launch(void* const* d_in, const int* in_sizes, int n_in,
                              void* d_out, int out_size, void* d_ws, size_t ws_size,
                              hipStream_t stream) {
  const float* q    = (const float*)d_in[0];
  const float* k    = (const float*)d_in[1];
  const float* v    = (const float*)d_in[2];
  const int*   rows = (const int*)d_in[3];
  const int*   cols = (const int*)d_in[4];
  const int    nnz  = in_sizes[3];

  dim3 grid(NB_, B_ * H_);   // 128 block-rows x 32 (batch*head)
  bsattn_wmma_bf16<<<grid, 32, 0, stream>>>(q, k, v, rows, cols, (float*)d_out, nnz);
}
